// SignalRNN_13271448944869
// MI455X (gfx1250) — matrix operand order repaired and checked by reading, not compiled, not run on hardware
//
#include <hip/hip_runtime.h>

typedef __attribute__((ext_vector_type(16))) __bf16 v16bf;
typedef __attribute__((ext_vector_type(8)))  __bf16 v8bf;
typedef __attribute__((ext_vector_type(8)))  float  v8f;

#define RNN_BATCH 512
#define RNN_IN    64
#define RNN_HID   256
#define RNN_OUT   64
#define MT        16           // batch rows per workgroup (WMMA M)
#define NWAVE_REC 16           // recurrence waves: one 16-col hidden tile each
#define NWAVE_PRJ 4            // projection waves: one 16-col output tile each
#define NTHREADS  ((NWAVE_REC + NWAVE_PRJ) * 32)   // 640 threads = 20 waves

// Branch-free tanh on the v_exp_f32 / v_rcp_f32 hardware path:
// tanh(x) = 1 - 2/(2^(2*log2e*x) + 1); exact saturation at +/-1.
__device__ __forceinline__ float fast_tanh(float x) {
    float e = __builtin_amdgcn_exp2f(x * 2.88539008177793f);  // 2*log2(e)
    return 1.0f - 2.0f * __builtin_amdgcn_rcpf(e + 1.0f);
}

// Load one 16x32 bf16 A-fragment from LDS row-major h state.
// Lane<16: K {0..7, 16..23}+32c ; lane>=16: K {8..15, 24..31}+32c.
__device__ __forceinline__ v16bf load_a_frag(const __bf16* rowp, int kb) {
    v8bf a0 = *(const v8bf*)(rowp + kb);
    v8bf a1 = *(const v8bf*)(rowp + kb + 16);
    v16bf a;
    #pragma unroll
    for (int i = 0; i < 8; ++i) { a[i] = a0[i]; a[i + 8] = a1[i]; }
    return a;
}

__global__ __launch_bounds__(NTHREADS, 1)
void rnn_fused_kernel(const float* __restrict__ x,
                      const float* __restrict__ W_ih, const float* __restrict__ b_ih,
                      const float* __restrict__ W_hh, const float* __restrict__ b_hh,
                      const float* __restrict__ W_out, const float* __restrict__ b_out,
                      const int S,
                      float* __restrict__ out,      // [B, S, OUT]
                      float* __restrict__ hidden)   // [B, HID]
{
    const int tid  = threadIdx.x;
    const int w    = tid >> 5;          // wave id (wave32)
    const int lane = tid & 31;
    const int l16  = lane & 15;
    const int hi   = lane >> 4;         // lane half: 0 or 1
    const int b0   = blockIdx.x * MT;   // batch tile base row

    // Double-buffered hidden state in bf16, row-major [M][HID], A-fragment ready.
    __shared__ __attribute__((aligned(16))) __bf16 hA[2][MT][RNN_HID];

    // ---- zero hA[0] (h0 = 0); re-done every launch -> deterministic ----
    {
        uint32_t* p = (uint32_t*)&hA[0][0][0];
        const int ndw = MT * RNN_HID / 2;           // 2048 dwords
        for (int i = tid; i < ndw; i += NTHREADS) p[i] = 0u;
    }

    // ---- persistent per-wave state -------------------------------------
    // B fragments (32x16 bf16 per K-chunk): lane n holds column N=n (l16);
    // lanes 0-15 carry K {0..7, 16..23}+32c, lanes 16-31 carry K {8..15, 24..31}+32c.
    v16bf Bfrag[8];
    v8f   P  = {};      // recurrence waves: xproj + b_ih + b_hh in C/D layout
    float bo = 0.0f;    // projection waves: output bias
    float hv[8];        // recurrence waves: last hidden values (f32)
    #pragma unroll
    for (int r = 0; r < 8; ++r) hv[r] = 0.0f;

    if (w < NWAVE_REC) {
        const int j = w * 16 + l16;                 // hidden column N owned by lane
        #pragma unroll
        for (int c = 0; c < 8; ++c) {               // K chunks of 32
            const int kb = c * 32 + hi * 8;
            const float* src = W_hh + (size_t)j * RNN_HID + kb;  // B[k][n] = W_hh[n][k]
            v16bf f;
            #pragma unroll
            for (int i = 0; i < 16; ++i) f[i] = (__bf16)src[(i & 7) + ((i >> 3) << 4)];
            Bfrag[c] = f;
        }
        // P in C/D layout: VGPR r -> row M = r + hi*8, column j.
        const float bias = b_ih[j] + b_hh[j];
        #pragma unroll
        for (int r = 0; r < 8; ++r) {
            const int m = r + hi * 8;
            const float* xr = x    + (size_t)(b0 + m) * RNN_IN;
            const float* wr = W_ih + (size_t)j * RNN_IN;
            float acc = bias;
            for (int k = 0; k < RNN_IN; ++k) acc += xr[k] * wr[k];
            P[r] = acc;
        }
    } else {
        const int j = (w - NWAVE_REC) * 16 + l16;   // output column (0..63)
        #pragma unroll
        for (int c = 0; c < 8; ++c) {
            const int kb = c * 32 + hi * 8;
            const float* src = W_out + (size_t)j * RNN_HID + kb; // B[k][o] = W_out[o][k]
            v16bf f;
            #pragma unroll
            for (int i = 0; i < 16; ++i) f[i] = (__bf16)src[(i & 7) + ((i >> 3) << 4)];
            Bfrag[c] = f;
        }
        bo = b_out[j];
    }

    __syncthreads();

    // ---- main loop: iteration t consumes h_t, produces h_{t+1};
    //      projection waves emit out[t-1] from h_t (shifted), epilogue at t==S.
    int cur = 0;
    const int m_a = l16;    // A-fragment row this lane loads (M = lane&15)

    for (int t = 0; t <= S; ++t) {
        if (w < NWAVE_REC) {
            if (t < S) {
                // Preload ALL 8 A fragments -> 16 ds_load_b128 issue together,
                // then 8 chained WMMAs (single dscnt wait amortized per step).
                const __bf16* rowp = &hA[cur][m_a][0];
                v16bf afr[8];
                #pragma unroll
                for (int c = 0; c < 8; ++c) afr[c] = load_a_frag(rowp, c * 32 + hi * 8);

                v8f acc = {};
                #pragma unroll
                for (int c = 0; c < 8; ++c)
                    acc = __builtin_amdgcn_wmma_f32_16x16x32_bf16(
                              false, afr[c], false, Bfrag[c], (short)0, acc, false, false);

                const int nxt = cur ^ 1;
                const int jn  = w * 16 + l16;
                #pragma unroll
                for (int r = 0; r < 8; ++r) {
                    const float h = fast_tanh(P[r] + acc[r]);
                    hv[r] = h;
                    hA[nxt][r + hi * 8][jn] = (__bf16)h;
                }
            }
        } else {
            if (t >= 1) {
                const __bf16* rowp = &hA[cur][m_a][0];
                v16bf afr[8];
                #pragma unroll
                for (int c = 0; c < 8; ++c) afr[c] = load_a_frag(rowp, c * 32 + hi * 8);

                v8f acc = {};
                #pragma unroll
                for (int c = 0; c < 8; ++c)
                    acc = __builtin_amdgcn_wmma_f32_16x16x32_bf16(
                              false, afr[c], false, Bfrag[c], (short)0, acc, false, false);

                const int jo = (w - NWAVE_REC) * 16 + l16;
                const int tt = t - 1;
                #pragma unroll
                for (int r = 0; r < 8; ++r) {
                    const int m = r + hi * 8;
                    out[((size_t)(b0 + m) * S + tt) * RNN_OUT + jo] = acc[r] + bo;
                }
            }
        }
        __syncthreads();
        cur ^= 1;
    }

    // ---- final hidden state [1, B, HID] -------------------------------
    if (w < NWAVE_REC) {
        const int jn = w * 16 + l16;
        #pragma unroll
        for (int r = 0; r < 8; ++r)
            hidden[(size_t)(b0 + r + hi * 8) * RNN_HID + jn] = hv[r];
    }
}

extern "C" void kernel_launch(void* const* d_in, const int* in_sizes, int n_in,
                              void* d_out, int out_size, void* d_ws, size_t ws_size,
                              hipStream_t stream) {
    const float* x     = (const float*)d_in[0];
    const float* W_ih  = (const float*)d_in[1];
    const float* b_ih  = (const float*)d_in[2];
    const float* W_hh  = (const float*)d_in[3];
    const float* b_hh  = (const float*)d_in[4];
    const float* W_out = (const float*)d_in[5];
    const float* b_out = (const float*)d_in[6];
    (void)in_sizes; (void)n_in; (void)d_ws; (void)ws_size;

    // d_out = [B,S,OUT] flat ++ [1,B,HID] flat  ->  recover S from out_size.
    const int S = (out_size - RNN_BATCH * RNN_HID) / (RNN_BATCH * RNN_OUT);

    float* out    = (float*)d_out;
    float* hidden = out + (size_t)RNN_BATCH * S * RNN_OUT;

    dim3 grid(RNN_BATCH / MT);   // 32 workgroups, one per 16-row batch tile
    dim3 block(NTHREADS);        // 640 threads = 20 wave32 waves
    rnn_fused_kernel<<<grid, block, 0, stream>>>(x, W_ih, b_ih, W_hh, b_hh,
                                                 W_out, b_out, S, out, hidden);
}
